// MaskedConvLayer_27341761806837
// MI455X (gfx1250) — matrix operand order repaired and checked by reading, not compiled, not run on hardware
//
#include <hip/hip_runtime.h>
#include <hip/hip_bf16.h>
#include <math.h>

typedef __attribute__((ext_vector_type(16))) _Float16 v16h;
typedef __attribute__((ext_vector_type(8)))  _Float16 v8h;
typedef __attribute__((ext_vector_type(4)))  _Float16 v4h;
typedef __attribute__((ext_vector_type(8)))  float    v8f;

#define F_DIM   128   // atom feature len
#define K_DIM   64    // neighbor feature len
#define M_NBR   12    // real neighbors (padded to 16 rows per tile)
#define IN_DIM  320   // 2F + K
#define OUT_DIM 256   // 2F
#define BN_EPS  1e-5f
#define APB     4     // atoms (row-tiles) per block: B-fragment reuse factor

// -------- workspace layout (bytes) --------
#define WS_WHALF   0           // 256*320 f16 = 163840
#define WS_SUM1    163840      // 256 f32
#define WS_SS1     164864      // 256 f32
#define WS_SCALE1  165888      // 256 f32
#define WS_SHIFT1  166912      // 256 f32
#define WS_SUM2    167936      // 128 f32
#define WS_SS2     168448      // 128 f32
#define WS_SCALE2  168960      // 128 f32
#define WS_SHIFT2  169472      // 128 f32
#define WS_NSUM    262144      // N*128 f32 (~25.6 MB)

// ---------------- prep: W f32 -> f16, zero stat accumulators ----------------
__global__ void mcl_prep(const float* __restrict__ W, _Float16* __restrict__ Wh,
                         float* __restrict__ s1, float* __restrict__ s2) {
    int tid = blockIdx.x * 256 + threadIdx.x;
    if (tid < OUT_DIM * IN_DIM) Wh[tid] = (_Float16)W[tid];
    if (tid < 512) s1[tid] = 0.f;   // sum1[256] ++ ss1[256] (contiguous)
    if (tid < 256) s2[tid] = 0.f;   // sum2[128] ++ ss2[128] (contiguous)
}

// ---------------- WMMA fragment helpers (ISA 7.12.2 layouts) ----------------
// A 16x32 f16: lane -> row = lane&15, half = lane>>4.
// per-lane elements: [8 halves at kbase+half*8] ++ [8 halves at kbase+16+half*8]
__device__ __forceinline__ v16h load_a_frag(const _Float16* __restrict__ Alds,
                                            int kbase, int lane) {
    const int row  = lane & 15;
    const int half = lane >> 4;
    const _Float16* p = Alds + row * IN_DIM + kbase + half * 8;
    v8h lo = *(const v8h*)(p);
    v8h hi = *(const v8h*)(p + 16);
    return __builtin_shufflevector(lo, hi, 0,1,2,3,4,5,6,7,8,9,10,11,12,13,14,15);
}

// B 32x16 f16: lane -> col n = lane&15, half = lane>>4; elements =
// [8 K at kbase+half*8] ++ [8 K at kbase+16+half*8].  B[k][n] = W[n][k] (row-major W).
__device__ __forceinline__ v16h load_b_frag(const _Float16* __restrict__ Wh,
                                            int tile, int kbase, int lane) {
    const int n    = lane & 15;
    const int half = lane >> 4;
    const _Float16* p = Wh + (size_t)(tile * 16 + n) * IN_DIM + kbase + half * 8;
    v8h lo = *(const v8h*)(p);
    v8h hi = *(const v8h*)(p + 16);
    return __builtin_shufflevector(lo, hi, 0,1,2,3,4,5,6,7,8,9,10,11,12,13,14,15);
}

// ---------------- cooperative A staging: APB atoms' 16x320 masked f16 tiles ----------------
__device__ __forceinline__ void stage_tiles(const float* __restrict__ atom_in,
                                            const float* __restrict__ nbr_fea,
                                            const int* __restrict__ nbr_idx,
                                            int i0, int nAtoms, _Float16* Alds,
                                            int* sidx, float* smask) {
    if (threadIdx.x < APB * 16) {
        int a = threadIdx.x >> 4, j = threadIdx.x & 15;
        int ii = i0 + a;
        int iq = (ii < nAtoms) ? ii : 0;
        int id = (j < M_NBR) ? nbr_idx[(size_t)iq * M_NBR + j] : 0;
        sidx[threadIdx.x]  = (j < M_NBR) ? id : 0;
        smask[threadIdx.x] = (j < M_NBR && id != 0 && ii < nAtoms) ? 1.f : 0.f;
    }
    __syncthreads();
    // APB*16 rows * 80 float4-chunks = APB*1280 chunks, 128 threads
#pragma unroll 4
    for (int it = 0; it < APB * 10; ++it) {
        int e   = it * 128 + threadIdx.x;
        int a   = e / 1280;
        int rem = e - a * 1280;
        int j   = rem / 80, q = rem - (rem / 80) * 80;
        int ii  = i0 + a;
        int iq  = (ii < nAtoms) ? ii : 0;
        const float* src; float m;
        if (q < 32)      { src = atom_in + (size_t)iq * F_DIM + q * 4;                    m = 1.f; }              // self
        else if (q < 64) { src = atom_in + (size_t)sidx[a*16+j] * F_DIM + (q - 32) * 4;   m = smask[a*16+j]; }    // gathered nbr
        else { int jj = (j < M_NBR) ? j : 0;
               src = nbr_fea + ((size_t)iq * M_NBR + jj) * K_DIM + (q - 64) * 4;          m = smask[a*16+j]; }    // edge fea
        float4 d = *(const float4*)src;
        v4h h = { (_Float16)(d.x * m), (_Float16)(d.y * m),
                  (_Float16)(d.z * m), (_Float16)(d.w * m) };
        *(v4h*)(Alds + (size_t)(a * 16 + j) * IN_DIM + q * 4) = h;
    }
    __syncthreads();
}

#define WMMA(A, B, C) __builtin_amdgcn_wmma_f32_16x16x32_f16(false, (A), false, (B), (short)0, (C), false, false)

// core GEMM micro-kernel: 4 col-tiles (B in regs) x APB row-tiles
#define GEMM_LOOP(ACC)                                                         \
    _Pragma("unroll")                                                          \
    for (int kc = 0; kc < 10; ++kc) {                                          \
        const int kb = kc * 32;                                                \
        v16h b0 = load_b_frag(Wh, t0, kb, lane);                               \
        v16h b1 = load_b_frag(Wh, t1, kb, lane);                               \
        v16h b2 = load_b_frag(Wh, t2, kb, lane);                               \
        v16h b3 = load_b_frag(Wh, t3, kb, lane);                               \
        _Pragma("unroll")                                                      \
        for (int rt = 0; rt < APB; ++rt) {                                     \
            v16h a = load_a_frag(Alds + (size_t)rt * 16 * IN_DIM, kb, lane);   \
            ACC[rt][0] = WMMA(a, b0, ACC[rt][0]);                              \
            ACC[rt][1] = WMMA(a, b1, ACC[rt][1]);                              \
            ACC[rt][2] = WMMA(a, b2, ACC[rt][2]);                              \
            ACC[rt][3] = WMMA(a, b3, ACC[rt][3]);                              \
        }                                                                      \
    }

// ---------------- pass 1: WMMA GEMM + BN1 sum/sumsq (real rows only) ----------------
__global__ void __launch_bounds__(128)
mcl_gemm_stats(const float* __restrict__ atom_in, const float* __restrict__ nbr_fea,
               const int* __restrict__ nbr_idx, const _Float16* __restrict__ Wh,
               const float* __restrict__ bias, int nAtoms,
               float* __restrict__ sum1, float* __restrict__ ss1) {
    __shared__ _Float16 Alds[APB * 16 * IN_DIM];
    __shared__ int   sidx[APB * 16];
    __shared__ float smask[APB * 16];
    const int i0 = blockIdx.x * APB;
    stage_tiles(atom_in, nbr_fea, nbr_idx, i0, nAtoms, Alds, sidx, smask);

    const int lane = threadIdx.x & 31;
    const int wave = threadIdx.x >> 5;
    const int t0 = wave, t1 = wave + 4, t2 = wave + 8, t3 = wave + 12;
    v8f acc[APB][4] = {};
    GEMM_LOOP(acc)

    const int half = lane >> 4, n = lane & 15;
#pragma unroll
    for (int rt = 0; rt < APB; ++rt) {
        const float rowok = (i0 + rt < nAtoms) ? 1.f : 0.f;
#pragma unroll
        for (int sl = 0; sl < 4; ++sl) {
            const int tile = wave + sl * 4;
            const int c = tile * 16 + n;
            const float bv = bias[c];
            float s = 0.f, ss = 0.f;
#pragma unroll
            for (int v = 0; v < 8; ++v) {
                float keep = ((half * 8 + v < M_NBR) ? 1.f : 0.f) * rowok;  // exclude pad rows / pad atoms
                float x = acc[rt][sl][v] + bv;
                s  += x * keep;
                ss += x * x * keep;
            }
            s  += __shfl_xor(s,  16, 32);
            ss += __shfl_xor(ss, 16, 32);
            if (lane < 16) { atomicAdd(&sum1[c], s); atomicAdd(&ss1[c], ss); }
        }
    }
}

// ---------------- finalize BN: scale/shift from sum/sumsq ----------------
__global__ void mcl_finalize(const float* __restrict__ sum, const float* __restrict__ ss,
                             const float* __restrict__ gamma, const float* __restrict__ beta,
                             float cnt, int nch,
                             float* __restrict__ scale, float* __restrict__ shift) {
    int c = blockIdx.x * blockDim.x + threadIdx.x;
    if (c < nch) {
        float mean = sum[c] / cnt;
        float var  = ss[c] / cnt - mean * mean;
        float sc   = gamma[c] * rsqrtf(var + BN_EPS);
        scale[c] = sc;
        shift[c] = beta[c] - mean * sc;
    }
}

// ---------------- pass 2: WMMA GEMM + BN1 affine + sigmoid*softplus*mask + nbr sum ----------------
__global__ void __launch_bounds__(128)
mcl_gemm_conv(const float* __restrict__ atom_in, const float* __restrict__ nbr_fea,
              const int* __restrict__ nbr_idx, const _Float16* __restrict__ Wh,
              const float* __restrict__ bias, int nAtoms,
              const float* __restrict__ scale1, const float* __restrict__ shift1,
              float* __restrict__ nsum) {
    __shared__ _Float16 Alds[APB * 16 * IN_DIM];
    __shared__ int   sidx[APB * 16];
    __shared__ float smask[APB * 16];
    const int i0 = blockIdx.x * APB;
    stage_tiles(atom_in, nbr_fea, nbr_idx, i0, nAtoms, Alds, sidx, smask);

    const int lane = threadIdx.x & 31;
    const int wave = threadIdx.x >> 5;
    // slots 0,1 = filter tiles {w, w+4}; slots 2,3 = core tiles {w+8, w+12} (cols +128)
    const int t0 = wave, t1 = wave + 4, t2 = wave + 8, t3 = wave + 12;
    v8f acc[APB][4] = {};
    GEMM_LOOP(acc)

    const int half = lane >> 4, n = lane & 15;
#pragma unroll
    for (int rt = 0; rt < APB; ++rt) {
        const int i = i0 + rt;
        float mrow[8];
#pragma unroll
        for (int v = 0; v < 8; ++v) mrow[v] = smask[rt * 16 + half * 8 + v];  // pad rows = 0

        auto finish = [&](const v8f& fac, const v8f& cac, int tf) {
            const int cf = tf * 16 + n, cc = cf + 128;
            const float sF = scale1[cf], hF = shift1[cf], bF = bias[cf];
            const float sC = scale1[cc], hC = shift1[cc], bC = bias[cc];
            float s = 0.f;
#pragma unroll
            for (int v = 0; v < 8; ++v) {
                float yf = (fac[v] + bF) * sF + hF;
                float yc = (cac[v] + bC) * sC + hC;
                float fl = 1.f / (1.f + expf(-yf));              // sigmoid
                float cp = (yc > 20.f) ? yc : log1pf(expf(yc));  // softplus
                s += fl * cp * mrow[v];
            }
            s += __shfl_xor(s, 16, 32);                          // combine row halves
            if (lane < 16 && i < nAtoms) nsum[(size_t)i * F_DIM + cf] = s;
        };
        finish(acc[rt][0], acc[rt][2], t0);
        finish(acc[rt][1], acc[rt][3], t1);
    }
}

// ---------------- BN2 stats over N rows ----------------
__global__ void mcl_stats2(const float* __restrict__ nsum, int n,
                           float* __restrict__ sum2, float* __restrict__ ss2) {
    const int c = threadIdx.x;                    // 128 channels
    float s = 0.f, ss = 0.f;
    for (int r = blockIdx.x; r < n; r += gridDim.x) {
        float x = nsum[(size_t)r * F_DIM + c];
        s += x; ss += x * x;
    }
    atomicAdd(&sum2[c], s);
    atomicAdd(&ss2[c], ss);
}

// ---------------- output: softplus(atom_in + BN2(nsum)) ----------------
__global__ void mcl_out(const float* __restrict__ atom_in, const float* __restrict__ nsum,
                        const float* __restrict__ scale2, const float* __restrict__ shift2,
                        float* __restrict__ out, int total) {
    int t = blockIdx.x * 256 + threadIdx.x;
    if (t < total) {
        int c = t & (F_DIM - 1);
        float y = atom_in[t] + nsum[t] * scale2[c] + shift2[c];
        out[t] = (y > 20.f) ? y : log1pf(expf(y));
    }
}

extern "C" void kernel_launch(void* const* d_in, const int* in_sizes, int n_in,
                              void* d_out, int out_size, void* d_ws, size_t ws_size,
                              hipStream_t stream) {
    const float* atom_in = (const float*)d_in[0];
    const float* nbr_fea = (const float*)d_in[1];
    const int*   nbr_idx = (const int*)d_in[2];
    const float* W       = (const float*)d_in[3];
    const float* bias    = (const float*)d_in[4];
    const float* gamma1  = (const float*)d_in[5];
    const float* beta1   = (const float*)d_in[6];
    const float* gamma2  = (const float*)d_in[7];
    const float* beta2   = (const float*)d_in[8];
    float* out = (float*)d_out;

    const int N = in_sizes[0] / F_DIM;
    const int nBlocks = (N + APB - 1) / APB;

    char* ws = (char*)d_ws;
    _Float16* Wh   = (_Float16*)(ws + WS_WHALF);
    float* sum1    = (float*)(ws + WS_SUM1);
    float* ss1     = (float*)(ws + WS_SS1);
    float* scale1  = (float*)(ws + WS_SCALE1);
    float* shift1  = (float*)(ws + WS_SHIFT1);
    float* sum2    = (float*)(ws + WS_SUM2);
    float* ss2     = (float*)(ws + WS_SS2);
    float* scale2  = (float*)(ws + WS_SCALE2);
    float* shift2  = (float*)(ws + WS_SHIFT2);
    float* nsum    = (float*)(ws + WS_NSUM);

    mcl_prep<<<(OUT_DIM * IN_DIM + 255) / 256, 256, 0, stream>>>(W, Wh, sum1, sum2);

    mcl_gemm_stats<<<nBlocks, 128, 0, stream>>>(atom_in, nbr_fea, nbr_idx, Wh, bias, N,
                                                sum1, ss1);

    mcl_finalize<<<1, 256, 0, stream>>>(sum1, ss1, gamma1, beta1,
                                        (float)N * (float)M_NBR, OUT_DIM, scale1, shift1);

    mcl_gemm_conv<<<nBlocks, 128, 0, stream>>>(atom_in, nbr_fea, nbr_idx, Wh, bias, N,
                                               scale1, shift1, nsum);

    mcl_stats2<<<256, 128, 0, stream>>>(nsum, N, sum2, ss2);

    mcl_finalize<<<1, 128, 0, stream>>>(sum2, ss2, gamma2, beta2,
                                        (float)N, F_DIM, scale2, shift2);

    mcl_out<<<(N * F_DIM + 255) / 256, 256, 0, stream>>>(atom_in, nsum, scale2, shift2,
                                                         out, N * F_DIM);
}
